// FinerPCN_48747878809833
// MI455X (gfx1250) — compile-verified, matrix operand order checked by reading, not run on hardware
//
#include <hip/hip_runtime.h>
#include <hip/hip_bf16.h>
#include <math.h>

typedef __attribute__((ext_vector_type(16))) _Float16 v16h;
typedef __attribute__((ext_vector_type(8)))  float    v8f;

#define BATCH 16
#define NPTS  8192
#define NCOARSE 512
#define NALL  (NCOARSE + NPTS)   // 8704
#define KFPS  1024
#define KNN   16
#define DENS_SIG2 0.1156f        // 0.34^2

// ---------------------------------------------------------------------------
// Generic WMMA GEMM:  Y = epilogue( W[O,C] @ X[C,Ncols] )
//   per-block tile: 128 (M) x 32 (N), K-step 32, 8 waves of 32 (wave32)
//   each wave: rows w*16..w*16+15, both 16-wide N halves -> 2 x v_wmma per step
//   mode 0: y = acc + bias[m]
//   mode 1: y = relu(scale[m]*(acc+bias[m]) + shift[m])     (conv+BN+ReLU)
//   mode 2: y = relu(acc + bias[m])                         (linear+ReLU)
//   xconcat > 0: rows k < xconcat of X are broadcast vector g[b][k]
//                rows k >= xconcat come from X[(k-xconcat)][n]
// Fast path (vectorized b128 loads, no per-element guards) when
//   C%32==0 && Ncols%32==0 && m0+128<=O  -- true for all heavy layers.
// ---------------------------------------------------------------------------
#define GM 128
#define GN 32
#define GK 32
#define APAD 4   // halves; keeps 4-half groups 8B aligned

__global__ __launch_bounds__(256)
void k_gemm(const float* __restrict__ W, const float* __restrict__ X,
            const float* __restrict__ g, const float* __restrict__ bias,
            const float* __restrict__ scale, const float* __restrict__ shift,
            float* __restrict__ Y,
            int O, int C, int Ncols, long xbstride, long ybstride,
            int mode, int xconcat)
{
    __shared__ _Float16 As[GM][GK + APAD];
    __shared__ _Float16 Bs[GN][GK + APAD];

    const int b   = blockIdx.z;
    const float* Xb = X + (long)b * xbstride;
    const float* gb = (xconcat > 0) ? (g + (long)b * xconcat) : nullptr;
    float* Yb = Y + (long)b * ybstride;

    const int m0 = blockIdx.x * GM;
    const int n0 = blockIdx.y * GN;
    const int tid  = threadIdx.x;
    const int lane = tid & 31;
    const int wave = tid >> 5;

    v8f acc0 = {0.f, 0.f, 0.f, 0.f, 0.f, 0.f, 0.f, 0.f};
    v8f acc1 = {0.f, 0.f, 0.f, 0.f, 0.f, 0.f, 0.f, 0.f};

    const bool fast = ((C & 31) == 0) && ((Ncols & 31) == 0) && (m0 + GM <= O);
    const int nk = (C + GK - 1) / GK;

    for (int kt = 0; kt < nk; ++kt) {
        const int k0 = kt * GK;
        if (fast) {
            // ---- A tile: 128x32 f32 = 1024 float4, 4 per thread ----
#pragma unroll
            for (int j = 0; j < 4; ++j) {
                int chunk = tid + j * 256;          // [0,1024)
                int r = chunk >> 3;                 // row 0..127
                int k = (chunk & 7) * 4;            // k 0,4,..,28
                const float4 v = *(const float4*)(W + (long)(m0 + r) * C + (k0 + k));
                _Float16* dst = &As[r][k];
                dst[0] = (_Float16)v.x; dst[1] = (_Float16)v.y;
                dst[2] = (_Float16)v.z; dst[3] = (_Float16)v.w;
                if (kt + 1 < nk)
                    __builtin_prefetch(W + (long)(m0 + r) * C + (k0 + GK + k), 0, 1);
            }
            // ---- B tile: 32(k) x 32(n) f32 = 256 float4, 1 per thread ----
            {
                int k = tid >> 3;                   // 0..31
                int n = (tid & 7) * 4;              // 0,4,..,28
                int gk = k0 + k;
                float4 v;
                if (xconcat > 0 && gk < xconcat) {
                    float s = gb[gk];
                    v.x = s; v.y = s; v.z = s; v.w = s;
                } else {
                    int kk = (xconcat > 0) ? (gk - xconcat) : gk;
                    v = *(const float4*)(Xb + (long)kk * Ncols + (n0 + n));
                    if (kt + 1 < nk && !(xconcat > 0 && gk + GK < xconcat))
                        __builtin_prefetch(Xb + (long)(kk + (xconcat > 0 ? 0 : GK)) * Ncols + (n0 + n), 0, 1);
                }
                Bs[n + 0][k] = (_Float16)v.x;
                Bs[n + 1][k] = (_Float16)v.y;
                Bs[n + 2][k] = (_Float16)v.z;
                Bs[n + 3][k] = (_Float16)v.w;
            }
        } else {
            // ---- guarded scalar loaders (C=6 layers, MLP Ncols=16) ----
            for (int i = tid; i < GM * GK; i += 256) {
                int r = i >> 5, k = i & 31;
                int gm = m0 + r, gk = k0 + k;
                float v = (gm < O && gk < C) ? W[(long)gm * C + gk] : 0.f;
                As[r][k] = (_Float16)v;
            }
            for (int i = tid; i < GN * GK; i += 256) {
                int k = i >> 5, n = i & 31;
                int gk = k0 + k, gn = n0 + n;
                float v = 0.f;
                if (gk < C && gn < Ncols) {
                    if (xconcat > 0 && gk < xconcat) v = gb[gk];
                    else {
                        int kk = (xconcat > 0) ? (gk - xconcat) : gk;
                        v = Xb[(long)kk * Ncols + gn];
                    }
                }
                Bs[n][k] = (_Float16)v;
            }
        }
        __syncthreads();

        // fragments: A 16x32 per wave; two B 32x16 fragments (N halves)
        const int mrow = wave * 16 + (lane & 15);
        const int kg   = (lane >> 4) * 8;
        const _Float16* ap  = &As[mrow][0];
        const _Float16* bp0 = &Bs[lane & 15][0];
        const _Float16* bp1 = &Bs[16 + (lane & 15)][0];
        v16h af, bf0, bf1;
#pragma unroll
        for (int p = 0; p < 4; ++p) {
            af[2*p]     = ap[kg + 2*p];
            af[2*p + 1] = ap[kg + 2*p + 1];
            af[8 + 2*p]     = ap[16 + kg + 2*p];
            af[8 + 2*p + 1] = ap[16 + kg + 2*p + 1];
            bf0[2*p]     = bp0[kg + 2*p];
            bf0[2*p + 1] = bp0[kg + 2*p + 1];
            bf0[8 + 2*p]     = bp0[16 + kg + 2*p];
            bf0[8 + 2*p + 1] = bp0[16 + kg + 2*p + 1];
            bf1[2*p]     = bp1[kg + 2*p];
            bf1[2*p + 1] = bp1[kg + 2*p + 1];
            bf1[8 + 2*p]     = bp1[16 + kg + 2*p];
            bf1[8 + 2*p + 1] = bp1[16 + kg + 2*p + 1];
        }
        acc0 = __builtin_amdgcn_wmma_f32_16x16x32_f16(
                   false, af, false, bf0, (short)0, acc0, false, false);
        acc1 = __builtin_amdgcn_wmma_f32_16x16x32_f16(
                   false, af, false, bf1, (short)0, acc1, false, false);
        __syncthreads();
    }

    // store; C/D layout: lane n = lane&15; VGPR r -> m = r + (lane>=16 ? 8 : 0)
    const int n  = n0 + (lane & 15);
    const int mbase = m0 + wave * 16 + ((lane >> 4) ? 8 : 0);
#pragma unroll
    for (int r = 0; r < 8; ++r) {
        int m = mbase + r;
        if (m < O) {
            if (n < Ncols) {
                float v = acc0[r] + bias[m];
                if (mode == 1)      v = fmaxf(scale[m] * v + shift[m], 0.f);
                else if (mode == 2) v = fmaxf(v, 0.f);
                Yb[(long)m * Ncols + n] = v;
            }
            if (n + 16 < Ncols) {
                float v = acc1[r] + bias[m];
                if (mode == 1)      v = fmaxf(scale[m] * v + shift[m], 0.f);
                else if (mode == 2) v = fmaxf(v, 0.f);
                Yb[(long)m * Ncols + n + 16] = v;
            }
        }
    }
}

// ---------------------------------------------------------------------------
// small helper kernels
// ---------------------------------------------------------------------------
__global__ void k_transpose_xyz(const float* __restrict__ in, float* __restrict__ out)
{   // [B,N,6] -> [B,6,N]
    long i = (long)blockIdx.x * blockDim.x + threadIdx.x;
    long tot = (long)BATCH * NPTS * 6;
    if (i >= tot) return;
    int b = (int)(i / (NPTS * 6));
    int r = (int)(i % (NPTS * 6));
    int n = r / 6, c = r % 6;
    out[(long)b * 6 * NPTS + (long)c * NPTS + n] = in[i];
}

__global__ void k_transpose_fg(const float* __restrict__ in, float* __restrict__ out)
{   // [16,1024] -> [1024,16]
    int i = blockIdx.x * blockDim.x + threadIdx.x;
    if (i >= BATCH * 1024) return;
    int b = i / 1024, c = i % 1024;
    out[c * BATCH + b] = in[i];
}

__global__ void k_rowmax(const float* __restrict__ X, float* __restrict__ out,
                         int Ncols, long bstride, int O)
{
    int b = blockIdx.y, o = blockIdx.x;
    const float* row = X + (long)b * bstride + (long)o * Ncols;
    float m = -INFINITY;
    for (int i = threadIdx.x; i < Ncols; i += blockDim.x) m = fmaxf(m, row[i]);
    __shared__ float sm[256];
    sm[threadIdx.x] = m;
    __syncthreads();
    for (int h = 128; h > 0; h >>= 1) {
        if (threadIdx.x < h) sm[threadIdx.x] = fmaxf(sm[threadIdx.x], sm[threadIdx.x + h]);
        __syncthreads();
    }
    if (threadIdx.x == 0) out[(long)b * O + o] = sm[0];
}

__global__ void k_write_coarse(const float* __restrict__ m3,   // [3072,16]
                               const float* __restrict__ xyz,  // [B,N,6]
                               float* __restrict__ outc)       // [B,8704,6]
{
    long i = (long)blockIdx.x * blockDim.x + threadIdx.x;
    long tot = (long)BATCH * NALL * 6;
    if (i >= tot) return;
    int b = (int)(i / (NALL * 6));
    int r = (int)(i % (NALL * 6));
    int p = r / 6, d = r % 6;
    float v;
    if (p < NCOARSE) v = m3[(p * 6 + d) * BATCH + b];
    else             v = xyz[((long)b * NPTS + (p - NCOARSE)) * 6 + d];
    outc[i] = v;
}

// Farthest-point sampling: one block per batch; matches jnp scan semantics
// (idx[0]=0; tie-break = lowest index, like jnp.argmax)
#define FPS_T 512
__global__ __launch_bounds__(FPS_T)
void k_fps(const float* __restrict__ pts,   // [B,8704,6] (coarse in d_out)
           float* __restrict__ dmin,        // [B,8704]
           float* __restrict__ c2,          // [B,1024,6]
           float* __restrict__ c2T)         // [B,6,1024]
{
    int b = blockIdx.x;
    const float* P = pts + (long)b * NALL * 6;
    float* dm = dmin + (long)b * NALL;
    __shared__ float lastp[6];
    __shared__ float redv[FPS_T];
    __shared__ int   redi[FPS_T];
    int tid = threadIdx.x;
    for (int i = tid; i < NALL; i += FPS_T) dm[i] = INFINITY;
    int last = 0;
    for (int s = 0; s < KFPS; ++s) {
        if (tid < 6) {
            float v = P[last * 6 + tid];
            lastp[tid] = v;
            c2[((long)b * KFPS + s) * 6 + tid] = v;
            c2T[(long)b * 6 * KFPS + tid * KFPS + s] = v;
        }
        __syncthreads();
        float bestv = -INFINITY; int besti = 0x7fffffff;
        for (int i = tid; i < NALL; i += FPS_T) {
            float d = 0.f;
#pragma unroll
            for (int c = 0; c < 6; ++c) { float t = P[i * 6 + c] - lastp[c]; d += t * t; }
            float nd = fminf(dm[i], d);
            dm[i] = nd;
            if (nd > bestv || (nd == bestv && i < besti)) { bestv = nd; besti = i; }
        }
        redv[tid] = bestv; redi[tid] = besti;
        __syncthreads();
        for (int h = FPS_T / 2; h > 0; h >>= 1) {
            if (tid < h) {
                float ov = redv[tid + h]; int oi = redi[tid + h];
                if (ov > redv[tid] || (ov == redv[tid] && oi < redi[tid])) {
                    redv[tid] = ov; redi[tid] = oi;
                }
            }
            __syncthreads();
        }
        last = redi[0];
        __syncthreads();
    }
}

__global__ void k_knn(const float* __restrict__ c2, int* __restrict__ knn)
{
    int b = blockIdx.y;
    int p = blockIdx.x * blockDim.x + threadIdx.x;
    if (p >= KFPS) return;
    const float* base = c2 + (long)b * KFPS * 6;
    float q[6];
#pragma unroll
    for (int c = 0; c < 6; ++c) q[c] = base[p * 6 + c];
    float bd[KNN]; int bi[KNN];
#pragma unroll
    for (int j = 0; j < KNN; ++j) { bd[j] = INFINITY; bi[j] = -1; }
    for (int i = 0; i < KFPS; ++i) {
        float d = 0.f;
#pragma unroll
        for (int c = 0; c < 6; ++c) { float t = base[i * 6 + c] - q[c]; d += t * t; }
        if (d < bd[KNN - 1]) {
            int j = KNN - 1;
            while (j > 0 && bd[j - 1] > d) { bd[j] = bd[j - 1]; bi[j] = bi[j - 1]; --j; }
            bd[j] = d; bi[j] = i;
        }
    }
    for (int j = 0; j < KNN; ++j) knn[((long)b * KFPS + p) * KNN + j] = bi[j];
}

__global__ void k_stats(const float* __restrict__ c2, const int* __restrict__ knn,
                        float* __restrict__ densT, float* __restrict__ lpT)
{
    int b = blockIdx.y;
    int p = blockIdx.x * blockDim.x + threadIdx.x;
    if (p >= KFPS) return;
    const float* base = c2 + (long)b * KFPS * 6;
    const int* kn = knn + ((long)b * KFPS + p) * KNN;
    float se[6] = {0, 0, 0, 0, 0, 0}, sm[6] = {0, 0, 0, 0, 0, 0};
    for (int k = 0; k < KNN; ++k) {
        const float* nb = base + kn[k] * 6;
#pragma unroll
        for (int c = 0; c < 6; ++c) {
            float v = nb[c];
            se[c] += expf(-v / DENS_SIG2);
            sm[c] += v;
        }
    }
#pragma unroll
    for (int c = 0; c < 6; ++c) {
        densT[(long)b * 6 * KFPS + c * KFPS + p] = se[c] * (1.f / KNN);
        lpT[(long)b * 6 * KFPS + c * KFPS + p]   = sm[c] * (1.f / KNN) - base[p * 6 + c];
    }
}

__global__ void k_foldreduce(const float* __restrict__ Fo, const float* __restrict__ D,
                             const float* __restrict__ L, float* __restrict__ fold)
{
    int b = blockIdx.y, c = blockIdx.x;
    long off = (long)b * 256 * KFPS + (long)c * KFPS;
    float s = 0.f;
    for (int i = threadIdx.x; i < KFPS; i += 256) s += Fo[off + i] * D[off + i] * L[off + i];
    __shared__ float sm[256];
    sm[threadIdx.x] = s;
    __syncthreads();
    for (int h = 128; h > 0; h >>= 1) {
        if (threadIdx.x < h) sm[threadIdx.x] += sm[threadIdx.x + h];
        __syncthreads();
    }
    if (threadIdx.x == 0) fold[b * 256 + c] = sm[0];
}

// dense layer on one column per batch (grid.x = batch, block >= O)
// mode 0: Wx+b ; mode 1: relu(s*(Wx+b)+sh) ; optional elementwise-max merge
__global__ void k_densecol(const float* __restrict__ x, const float* __restrict__ W,
                           const float* __restrict__ bias, const float* __restrict__ s,
                           const float* __restrict__ sh, const float* __restrict__ merge,
                           float* __restrict__ y,
                           int O, int C, int mode, int ystride, int yoff)
{
    int b = blockIdx.x, o = threadIdx.x;
    if (o >= O) return;
    const float* xv = x + (long)b * C;
    const float* w = W + (long)o * C;
    float a = 0.f;
    for (int c = 0; c < C; ++c) a += w[c] * xv[c];
    a += bias[o];
    if (mode == 1) a = fmaxf(s[o] * a + sh[o], 0.f);
    if (merge) a = fmaxf(a, merge[(long)b * O + o]);
    y[(long)b * ystride + yoff + o] = a;
}

__global__ void k_write_fine(const float* __restrict__ o48, float* __restrict__ outf)
{
    long i = (long)blockIdx.x * blockDim.x + threadIdx.x;
    long tot = (long)BATCH * NPTS * 6;     // 49152 per batch
    if (i >= tot) return;
    int b = (int)(i / (NPTS * 6));
    int f = (int)(i % (NPTS * 6));
    outf[i] = o48[b * 48 + (f >> 10)];     // all 1024 cfin columns identical
}

// ---------------------------------------------------------------------------
extern "C" void kernel_launch(void* const* d_in, const int* in_sizes, int n_in,
                              void* d_out, int out_size, void* d_ws, size_t ws_size,
                              hipStream_t stream)
{
    (void)in_sizes; (void)n_in; (void)out_size; (void)ws_size;
    // input order = setup_inputs() dict insertion order
    const float* xyz      = (const float*)d_in[0];
    const float* c256_w1  = (const float*)d_in[1];
    const float* c256_b1  = (const float*)d_in[2];
    const float* c256_s1  = (const float*)d_in[3];
    const float* c256_sh1 = (const float*)d_in[4];
    const float* c256_w2  = (const float*)d_in[5];
    const float* c256_b2  = (const float*)d_in[6];
    const float* c512_w   = (const float*)d_in[7];
    const float* c512_b   = (const float*)d_in[8];
    const float* c512_s   = (const float*)d_in[9];
    const float* c512_sh  = (const float*)d_in[10];
    const float* c1k_w1   = (const float*)d_in[11];
    const float* c1k_b1   = (const float*)d_in[12];
    const float* c1k_s1   = (const float*)d_in[13];
    const float* c1k_sh1  = (const float*)d_in[14];
    const float* c1k_w2   = (const float*)d_in[15];
    const float* c1k_b2   = (const float*)d_in[16];
    const float* cf_w1    = (const float*)d_in[17];
    const float* cf_b1    = (const float*)d_in[18];
    const float* cf_s1    = (const float*)d_in[19];
    const float* cf_sh1   = (const float*)d_in[20];
    const float* cf_w2    = (const float*)d_in[21];
    const float* cf_b2    = (const float*)d_in[22];
    const float* cf_s2    = (const float*)d_in[23];
    const float* cf_sh2   = (const float*)d_in[24];
    const float* cf_w3    = (const float*)d_in[25];
    const float* cf_b3    = (const float*)d_in[26];
    const float* mlp_W1   = (const float*)d_in[27];
    const float* mlp_b1   = (const float*)d_in[28];
    const float* mlp_W2   = (const float*)d_in[29];
    const float* mlp_b2   = (const float*)d_in[30];
    const float* mlp_W3   = (const float*)d_in[31];
    const float* mlp_b3   = (const float*)d_in[32];

    float* outc = (float*)d_out;                                    // [16,8704,6]
    float* outf = outc + (long)BATCH * NALL * 6;                    // [16,8192,6]

    // workspace carve-up (floats)
    float* w = (float*)d_ws;
    auto take = [&](long nflt) { float* p = w; w += nflt; return p; };
    float* xyzT   = take((long)BATCH * 6 * NPTS);
    float* f1     = take((long)BATCH * 128 * NPTS);
    float* f2     = take((long)BATCH * 256 * NPTS);
    float* hbuf   = take((long)BATCH * 512 * NPTS);
    float* f3     = take((long)BATCH * 1024 * NPTS);
    float* fg256  = take(BATCH * 256);
    float* fg1024 = take(BATCH * 1024);
    float* gT     = take(1024 * BATCH);
    float* m1     = take(1024 * BATCH);
    float* m2     = take(1024 * BATCH);
    float* m3     = take(3072 * BATCH);
    float* dmin   = take((long)BATCH * NALL);
    float* c2     = take((long)BATCH * KFPS * 6);
    float* c2T    = take((long)BATCH * 6 * KFPS);
    int*   knn    = (int*)take((long)BATCH * KFPS * KNN);
    float* densT  = take((long)BATCH * 6 * KFPS);
    float* lpT    = take((long)BATCH * 6 * KFPS);
    float* t128   = take((long)BATCH * 128 * KFPS);
    float* Dm     = take((long)BATCH * 256 * KFPS);
    float* Lm     = take((long)BATCH * 256 * KFPS);
    float* Fo     = take((long)BATCH * 256 * KFPS);
    float* fold256= take(BATCH * 256);
    float* fgf    = take(BATCH * 256);
    float* y512   = take((long)BATCH * 512 * KFPS);
    float* fmax   = take(BATCH * 512);
    float* vcat   = take(BATCH * 1024);
    float* h1     = take(BATCH * 512);
    float* h2     = take(BATCH * 256);
    float* o48    = take(BATCH * 48);

    // ---- global encoder ----
    k_transpose_xyz<<<(BATCH * NPTS * 6 + 255) / 256, 256, 0, stream>>>(xyz, xyzT);

    k_gemm<<<dim3(1, NPTS / 32, BATCH), 256, 0, stream>>>(
        c256_w1, xyzT, nullptr, c256_b1, c256_s1, c256_sh1, f1,
        128, 6, NPTS, (long)6 * NPTS, (long)128 * NPTS, 1, 0);
    k_gemm<<<dim3(2, NPTS / 32, BATCH), 256, 0, stream>>>(
        c256_w2, f1, nullptr, c256_b2, nullptr, nullptr, f2,
        256, 128, NPTS, (long)128 * NPTS, (long)256 * NPTS, 0, 0);
    k_rowmax<<<dim3(256, BATCH), 256, 0, stream>>>(f2, fg256, NPTS, (long)256 * NPTS, 256);

    // c1024 on virtual concat [fg256 ; f2]
    k_gemm<<<dim3(4, NPTS / 32, BATCH), 256, 0, stream>>>(
        c1k_w1, f2, fg256, c1k_b1, c1k_s1, c1k_sh1, hbuf,
        512, 512, NPTS, (long)256 * NPTS, (long)512 * NPTS, 1, 256);
    k_gemm<<<dim3(8, NPTS / 32, BATCH), 256, 0, stream>>>(
        c1k_w2, hbuf, nullptr, c1k_b2, nullptr, nullptr, f3,
        1024, 512, NPTS, (long)512 * NPTS, (long)1024 * NPTS, 0, 0);
    k_rowmax<<<dim3(1024, BATCH), 256, 0, stream>>>(f3, fg1024, NPTS, (long)1024 * NPTS, 1024);

    // ---- coarse MLP (batch folded into the 16 columns) ----
    k_transpose_fg<<<(BATCH * 1024 + 255) / 256, 256, 0, stream>>>(fg1024, gT);
    k_gemm<<<dim3(8, 1, 1), 256, 0, stream>>>(
        mlp_W1, gT, nullptr, mlp_b1, nullptr, nullptr, m1,
        1024, 1024, BATCH, 0, 0, 2, 0);
    k_gemm<<<dim3(8, 1, 1), 256, 0, stream>>>(
        mlp_W2, m1, nullptr, mlp_b2, nullptr, nullptr, m2,
        1024, 1024, BATCH, 0, 0, 2, 0);
    k_gemm<<<dim3(24, 1, 1), 256, 0, stream>>>(
        mlp_W3, m2, nullptr, mlp_b3, nullptr, nullptr, m3,
        3072, 1024, BATCH, 0, 0, 0, 0);

    k_write_coarse<<<((long)BATCH * NALL * 6 + 255) / 256, 256, 0, stream>>>(m3, xyz, outc);

    // ---- FPS + KNN geometry ----
    k_fps<<<BATCH, FPS_T, 0, stream>>>(outc, dmin, c2, c2T);
    k_knn<<<dim3(KFPS / 256, BATCH), 256, 0, stream>>>(c2, knn);
    k_stats<<<dim3(KFPS / 256, BATCH), 256, 0, stream>>>(c2, knn, densT, lpT);

    // three _c256 applications (density, lp, c2)
    k_gemm<<<dim3(1, KFPS / 32, BATCH), 256, 0, stream>>>(
        c256_w1, densT, nullptr, c256_b1, c256_s1, c256_sh1, t128,
        128, 6, KFPS, (long)6 * KFPS, (long)128 * KFPS, 1, 0);
    k_gemm<<<dim3(2, KFPS / 32, BATCH), 256, 0, stream>>>(
        c256_w2, t128, nullptr, c256_b2, nullptr, nullptr, Dm,
        256, 128, KFPS, (long)128 * KFPS, (long)256 * KFPS, 0, 0);
    k_gemm<<<dim3(1, KFPS / 32, BATCH), 256, 0, stream>>>(
        c256_w1, lpT, nullptr, c256_b1, c256_s1, c256_sh1, t128,
        128, 6, KFPS, (long)6 * KFPS, (long)128 * KFPS, 1, 0);
    k_gemm<<<dim3(2, KFPS / 32, BATCH), 256, 0, stream>>>(
        c256_w2, t128, nullptr, c256_b2, nullptr, nullptr, Lm,
        256, 128, KFPS, (long)128 * KFPS, (long)256 * KFPS, 0, 0);
    k_gemm<<<dim3(1, KFPS / 32, BATCH), 256, 0, stream>>>(
        c256_w1, c2T, nullptr, c256_b1, c256_s1, c256_sh1, t128,
        128, 6, KFPS, (long)6 * KFPS, (long)128 * KFPS, 1, 0);
    k_gemm<<<dim3(2, KFPS / 32, BATCH), 256, 0, stream>>>(
        c256_w2, t128, nullptr, c256_b2, nullptr, nullptr, Fo,
        256, 128, KFPS, (long)128 * KFPS, (long)256 * KFPS, 0, 0);

    k_foldreduce<<<dim3(256, BATCH), 256, 0, stream>>>(Fo, Dm, Lm, fold256);
    k_rowmax<<<dim3(256, BATCH), 256, 0, stream>>>(Fo, fgf, KFPS, (long)256 * KFPS, 256);

    // feat branch: c512 over Fo columns + the broadcast-fg column, then max
    k_gemm<<<dim3(4, KFPS / 32, BATCH), 256, 0, stream>>>(
        c512_w, Fo, nullptr, c512_b, c512_s, c512_sh, y512,
        512, 256, KFPS, (long)256 * KFPS, (long)512 * KFPS, 1, 0);
    k_rowmax<<<dim3(512, BATCH), 256, 0, stream>>>(y512, fmax, KFPS, (long)512 * KFPS, 512);

    // vcat = [ c512(fold256) ; max(c512(fgf), fmax) ]
    k_densecol<<<BATCH, 512, 0, stream>>>(fold256, c512_w, c512_b, c512_s, c512_sh,
                                          nullptr, vcat, 512, 256, 1, 1024, 0);
    k_densecol<<<BATCH, 512, 0, stream>>>(fgf, c512_w, c512_b, c512_s, c512_sh,
                                          fmax, vcat, 512, 256, 1, 1024, 512);

    // cfin on the single (replicated) column
    k_densecol<<<BATCH, 512, 0, stream>>>(vcat, cf_w1, cf_b1, cf_s1, cf_sh1,
                                          nullptr, h1, 512, 1024, 1, 512, 0);
    k_densecol<<<BATCH, 256, 0, stream>>>(h1, cf_w2, cf_b2, cf_s2, cf_sh2,
                                          nullptr, h2, 256, 512, 1, 256, 0);
    k_densecol<<<BATCH, 64, 0, stream>>>(h2, cf_w3, cf_b3, nullptr, nullptr,
                                         nullptr, o48, 48, 256, 0, 48, 0);

    k_write_fine<<<((long)BATCH * NPTS * 6 + 255) / 256, 256, 0, stream>>>(o48, outf);
}
